// EnhancedEdgeScorer_34016140985094
// MI455X (gfx1250) — matrix operand order repaired and checked by reading, not compiled, not run on hardware
//
#include <hip/hip_runtime.h>

#define N_NODES 16384
#define M_CTX   32
#define HDIM    256
#define NLAYERS 3
#define NEDGES  65536
#define TOTF    320

typedef __attribute__((ext_vector_type(16))) __bf16        v16bf;
typedef __attribute__((ext_vector_type(8)))  float         v8f;
typedef __attribute__((ext_vector_type(8)))  unsigned int  v8u;

__device__ __forceinline__ float bf2f(unsigned short u) {
  unsigned int x = ((unsigned int)u) << 16;
  return __builtin_bit_cast(float, x);
}
__device__ __forceinline__ unsigned short f2bf(float f) {
  unsigned int u = __builtin_bit_cast(unsigned int, f);
  u += 0x7fffu + ((u >> 16) & 1u);          // round-to-nearest-even
  return (unsigned short)(u >> 16);
}

// ---------------------------------------------------------------- converts
__global__ void conv_f32_bf16(const float* __restrict__ src,
                              unsigned short* __restrict__ dst, int n) {
  int i = blockIdx.x * blockDim.x + threadIdx.x;
  if (i < n) dst[i] = f2bf(src[i]);
}

// ---------------------------------------------------------------- encode F (N,320) bf16
__global__ void encode_feats(const int* __restrict__ type_idx,
                             const int* __restrict__ cat_idx,
                             const float* __restrict__ log_degree,
                             const float* __restrict__ type_embed,
                             const float* __restrict__ cat0,
                             const float* __restrict__ cat1,
                             const float* __restrict__ dw,
                             const float* __restrict__ db,
                             unsigned short* __restrict__ F) {
  int i = blockIdx.x * blockDim.x + threadIdx.x;
  if (i >= N_NODES * TOTF) return;
  int n = i / TOTF, j = i - n * TOTF;
  float v;
  if (j < 128)       v = type_embed[type_idx[n] * 128 + j];
  else if (j < 192)  v = cat0[cat_idx[n * 2 + 0] * 64 + (j - 128)];
  else if (j < 256)  v = cat1[cat_idx[n * 2 + 1] * 64 + (j - 192)];
  else               v = fmaxf(log_degree[n] * dw[j - 256] + db[j - 256], 0.f);
  F[i] = f2bf(v);
}

// ---------------------------------------------------------------- generic WMMA GEMM
// C(M x NBLK-slice) = act( A(M x K)[rowidx] @ W(Nout x K)^T + bias + Cinit )
// 256 threads = 8 waves, arranged WM(M) x WN(N); each wave owns a 16x64 strip
// (4 full 16x16 tiles -> no guards, EXEC stays all-ones around every WMMA).
// Weight slice (NBLK x K bf16, <=128KB) is LDS-resident (320KB/WGP budget).
template <int WN>
__global__ __launch_bounds__(256, 1)
void wmma_gemm(const unsigned short* __restrict__ A, int lda,
               const int* __restrict__ rowidx,
               const unsigned short* __restrict__ W, int ldw, int K,
               const float* __restrict__ bias,
               const float* __restrict__ Cinit, int ldc,
               unsigned short* __restrict__ outb, float* __restrict__ outf,
               int ldo, int relu) {
  constexpr int WM   = 8 / WN;
  constexpr int NBLK = WN * 64;
  __shared__ unsigned short sW[NBLK * 256];   // K <= 256

  const int nbase = blockIdx.y * NBLK;

  // stage weight slice into LDS once per block
  for (int idx = threadIdx.x; idx < NBLK * K; idx += 256) {
    int n = idx / K, k = idx - n * K;
    sW[idx] = W[(size_t)(nbase + n) * ldw + k];
  }
  __syncthreads();

  const int wave = threadIdx.x >> 5, lane = threadIdx.x & 31;
  const int wm = wave % WM, wn = wave / WM;
  const int lanelo = lane & 15, half = lane >> 4;
  const int rbase = blockIdx.x * (WM * 16) + wm * 16;
  const int arow  = rbase + lanelo;
  const int asrc  = rowidx ? rowidx[arow] : arow;
  const unsigned short* __restrict__ Aptr = A + (size_t)asrc * lda;

  // per-lane LDS base for this wave's 64-column strip:
  // B element (k,n): lanes 0-15 n=lane,k {0..7,16..23}; lanes 16-31 n=lane-16,k +8
  // fragment = 2x 4 consecutive dwords, 16B-aligned -> 2x ds_load_b128 per tile
  const unsigned short* __restrict__ B0 = sW + (size_t)(wn * 64 + lanelo) * K + 8 * half;

  v8f acc0 = {}, acc1 = {}, acc2 = {}, acc3 = {};
  for (int kb = 0; kb < K; kb += 32) {
    // A fragment 16x32: lanes 0-15 K {0..7,16..23}, lanes 16-31 +8
    v8u au;
#pragma unroll
    for (int r = 0; r < 8; ++r) {
      int k2 = 8 * half + (r < 4 ? 2 * r : 2 * r + 8);
      au[r] = *(const unsigned int*)(Aptr + kb + k2);
    }
    v16bf a = __builtin_bit_cast(v16bf, au);

    v8u bu[4];
#pragma unroll
    for (int t = 0; t < 4; ++t) {
      const uint4 lo = *(const uint4*)(B0 + t * 16 * K + kb);      // dwords 0..3
      const uint4 hi = *(const uint4*)(B0 + t * 16 * K + kb + 16); // dwords 4..7
      bu[t][0] = lo.x; bu[t][1] = lo.y; bu[t][2] = lo.z; bu[t][3] = lo.w;
      bu[t][4] = hi.x; bu[t][5] = hi.y; bu[t][6] = hi.z; bu[t][7] = hi.w;
    }
    acc0 = __builtin_amdgcn_wmma_f32_16x16x32_bf16(
        false, a, false, __builtin_bit_cast(v16bf, bu[0]), (short)0, acc0, false, false);
    acc1 = __builtin_amdgcn_wmma_f32_16x16x32_bf16(
        false, a, false, __builtin_bit_cast(v16bf, bu[1]), (short)0, acc1, false, false);
    acc2 = __builtin_amdgcn_wmma_f32_16x16x32_bf16(
        false, a, false, __builtin_bit_cast(v16bf, bu[2]), (short)0, acc2, false, false);
    acc3 = __builtin_amdgcn_wmma_f32_16x16x32_bf16(
        false, a, false, __builtin_bit_cast(v16bf, bu[3]), (short)0, acc3, false, false);
  }

  // epilogue: bias + init-accumulate + relu, write bf16 and/or fp32
  const v8f accs[4] = {acc0, acc1, acc2, acc3};
#pragma unroll
  for (int t = 0; t < 4; ++t) {
    const int n = nbase + wn * 64 + t * 16 + lanelo;
    const float bv = bias ? bias[n] : 0.f;
#pragma unroll
    for (int r = 0; r < 8; ++r) {
      const int row = rbase + r + 8 * half;
      float v = accs[t][r] + bv;
      if (Cinit) v += Cinit[(size_t)row * ldc + n];
      if (relu)  v = fmaxf(v, 0.f);
      if (outf)  outf[(size_t)row * (size_t)ldo + n] = v;
      if (outb)  outb[(size_t)row * (size_t)ldo + n] = f2bf(v);
    }
  }
}

// ---------------------------------------------------------------- attention (wave per node)
// qkv: (N,768) bf16 = [q|k|v]; gather k/v rows post-projection (algebraically
// identical to projecting the gathered context, 32x fewer k/v projection FLOPs).
__global__ __launch_bounds__(256)
void attn_ctx(const unsigned short* __restrict__ qkv,
              const int* __restrict__ ctx,
              const unsigned char* __restrict__ mask,
              unsigned short* __restrict__ o) {
  __shared__ float sq[8][256];
  __shared__ float swt[8][4][32];
  __shared__ int   sidx[8][32];

  const int wave = threadIdx.x >> 5, lane = threadIdx.x & 31;
  const int n = blockIdx.x * 8 + wave;

  const unsigned short* qrow = qkv + (size_t)n * 768;
  for (int j = lane; j < 256; j += 32) sq[wave][j] = bf2f(qrow[j]);
  const int cidx = ctx[n * 32 + lane];
  sidx[wave][lane] = cidx;
  const bool pad = mask[n * 32 + lane] != 0;
  __syncthreads();

  // lane <-> context slot; scores for 4 heads
  const unsigned short* krow = qkv + (size_t)cidx * 768 + 256;
  float sc[4];
#pragma unroll
  for (int h = 0; h < 4; ++h) {
    float part = 0.f;
    for (int d = h * 64; d < h * 64 + 64; d += 8) {
      uint4 kq = *(const uint4*)(krow + d);
      unsigned int ku[4] = {kq.x, kq.y, kq.z, kq.w};
#pragma unroll
      for (int j = 0; j < 4; ++j) {
        part += bf2f((unsigned short)(ku[j] & 0xffffu)) * sq[wave][d + 2 * j]
              + bf2f((unsigned short)(ku[j] >> 16))     * sq[wave][d + 2 * j + 1];
      }
    }
    sc[h] = part;
  }

  // masked softmax across the 32 context lanes (wave32 shuffles)
#pragma unroll
  for (int h = 0; h < 4; ++h) {
    float s = pad ? -1e30f : sc[h] * 0.125f;   // 1/sqrt(64)
    float mx = s;
    for (int off = 16; off > 0; off >>= 1) mx = fmaxf(mx, __shfl_xor(mx, off, 32));
    float e = __expf(s - mx);
    float sum = e;
    for (int off = 16; off > 0; off >>= 1) sum += __shfl_xor(sum, off, 32);
    swt[wave][h][lane] = e / sum;
  }
  __syncthreads();

  // weighted sum of V: lane owns 8 output dims
  const int d0 = lane * 8;
  const int hh = d0 >> 6;
  float acc[8] = {};
  for (int m = 0; m < 32; ++m) {
    float w = swt[wave][hh][m];
    const unsigned short* vrow = qkv + (size_t)sidx[wave][m] * 768 + 512 + d0;
    uint4 vq = *(const uint4*)vrow;
    unsigned int vu[4] = {vq.x, vq.y, vq.z, vq.w};
#pragma unroll
    for (int j = 0; j < 4; ++j) {
      acc[2 * j]     += w * bf2f((unsigned short)(vu[j] & 0xffffu));
      acc[2 * j + 1] += w * bf2f((unsigned short)(vu[j] >> 16));
    }
  }
  unsigned short* orow = o + (size_t)n * 256 + d0;
#pragma unroll
  for (int j = 0; j < 8; ++j) orow[j] = f2bf(acc[j]);
}

// ---------------------------------------------------------------- edge-feat rank-2 init: C0 = ef @ W1c^T + b1
__global__ void edge_c0(const float* __restrict__ ef,
                        const float* __restrict__ w1,
                        const float* __restrict__ b1,
                        float* __restrict__ C0) {
  int i = blockIdx.x * blockDim.x + threadIdx.x;
  if (i >= NEDGES * 256) return;
  int e = i >> 8, j = i & 255;
  C0[i] = ef[e * 2] * w1[(size_t)j * 514 + 512]
        + ef[e * 2 + 1] * w1[(size_t)j * 514 + 513] + b1[j];
}

// ---------------------------------------------------------------- final logits (Nout=1)
__global__ void edge_logits(const unsigned short* __restrict__ h2,
                            const float* __restrict__ w3,
                            const float* __restrict__ b3,
                            float* __restrict__ out) {
  int e = blockIdx.x * blockDim.x + threadIdx.x;
  if (e >= NEDGES) return;
  float acc = b3[0];
  const unsigned short* row = h2 + (size_t)e * 128;
#pragma unroll 8
  for (int k = 0; k < 128; ++k) acc += bf2f(row[k]) * w3[k];
  out[e] = acc;
}

// ================================================================ host
extern "C" void kernel_launch(void* const* d_in, const int* in_sizes, int n_in,
                              void* d_out, int out_size, void* d_ws, size_t ws_size,
                              hipStream_t stream) {
  (void)in_sizes; (void)n_in; (void)out_size; (void)ws_size;
  const int*   type_idx   = (const int*)d_in[0];
  const int*   cat_idx    = (const int*)d_in[1];
  const float* log_degree = (const float*)d_in[2];
  const int*   ctx_idx    = (const int*)d_in[3];
  const unsigned char* kpm = (const unsigned char*)d_in[4];
  const int*   u_idx      = (const int*)d_in[5];
  const int*   v_idx      = (const int*)d_in[6];
  const float* edge_feats = (const float*)d_in[7];
  const float* type_embed = (const float*)d_in[8];
  const float* cat0       = (const float*)d_in[9];
  const float* cat1       = (const float*)d_in[10];
  const float* dw         = (const float*)d_in[11];
  const float* db         = (const float*)d_in[12];
  const float* proj_w     = (const float*)d_in[13];
  const float* proj_b     = (const float*)d_in[14];
  const float* in_proj_w  = (const float*)d_in[15];
  const float* in_proj_b  = (const float*)d_in[16];
  const float* out_w      = (const float*)d_in[17];
  const float* out_b      = (const float*)d_in[18];
  const float* mlp_w1     = (const float*)d_in[19];
  const float* mlp_b1     = (const float*)d_in[20];
  const float* mlp_w2     = (const float*)d_in[21];
  const float* mlp_b2     = (const float*)d_in[22];
  const float* mlp_w3     = (const float*)d_in[23];
  const float* mlp_b3     = (const float*)d_in[24];
  float* logits = (float*)d_out;

  // workspace carve (256B aligned)
  char* p = (char*)d_ws;
  auto carve = [&](size_t bytes) -> char* {
    char* r = p; p += (bytes + 255) & ~(size_t)255; return r;
  };
  unsigned short* wbf_proj = (unsigned short*)carve((size_t)256 * 320 * 2);
  unsigned short* wbf_inp  = (unsigned short*)carve((size_t)3 * 768 * 256 * 2);
  unsigned short* wbf_out  = (unsigned short*)carve((size_t)3 * 256 * 256 * 2);
  unsigned short* wbf_m1   = (unsigned short*)carve((size_t)256 * 514 * 2);
  unsigned short* wbf_m2   = (unsigned short*)carve((size_t)128 * 256 * 2);
  unsigned short* F        = (unsigned short*)carve((size_t)N_NODES * TOTF * 2);
  unsigned short* xbuf     = (unsigned short*)carve((size_t)N_NODES * 256 * 2);
  unsigned short* qkv      = (unsigned short*)carve((size_t)N_NODES * 768 * 2);
  unsigned short* obuf     = (unsigned short*)carve((size_t)N_NODES * 256 * 2);
  float*          tmpf     = (float*)carve((size_t)NEDGES * 256 * 4);   // also covers N*256
  unsigned short* h1       = (unsigned short*)carve((size_t)NEDGES * 256 * 2);
  unsigned short* h2       = (unsigned short*)carve((size_t)NEDGES * 128 * 2);

  auto cvt = [&](const float* s, unsigned short* d, int cnt) {
    conv_f32_bf16<<<(cnt + 255) / 256, 256, 0, stream>>>(s, d, cnt);
  };
  // Nout multiple of 256 -> WN=4 (block 32x256); Nout==128 -> WN=2 (block 64x128)
  auto gemm = [&](const unsigned short* A, int lda, const int* ridx,
                  const unsigned short* W, int ldw, int K, int Nout,
                  const float* bias, const float* Ci, int ldc,
                  unsigned short* ob, float* of, int ldo, int relu, int Mrows) {
    if (Nout % 256 == 0) {
      dim3 g(Mrows / 32, Nout / 256);
      wmma_gemm<4><<<g, 256, 0, stream>>>(A, lda, ridx, W, ldw, K,
                                          bias, Ci, ldc, ob, of, ldo, relu);
    } else {
      dim3 g(Mrows / 64, Nout / 128);
      wmma_gemm<2><<<g, 256, 0, stream>>>(A, lda, ridx, W, ldw, K,
                                          bias, Ci, ldc, ob, of, ldo, relu);
    }
  };

  // weight conversion (bf16 for WMMA)
  cvt(proj_w,    wbf_proj, 256 * 320);
  cvt(in_proj_w, wbf_inp,  3 * 768 * 256);
  cvt(out_w,     wbf_out,  3 * 256 * 256);
  cvt(mlp_w1,    wbf_m1,   256 * 514);
  cvt(mlp_w2,    wbf_m2,   128 * 256);

  // node features
  encode_feats<<<(N_NODES * TOTF + 255) / 256, 256, 0, stream>>>(
      type_idx, cat_idx, log_degree, type_embed, cat0, cat1, dw, db, F);

  // x = F(320) @ proj_w^T + b : chained K = 256 + 64
  gemm(F, TOTF, nullptr, wbf_proj, 320, 256, 256, proj_b, nullptr, 0,
       nullptr, tmpf, 256, 0, N_NODES);
  gemm(F + 256, TOTF, nullptr, wbf_proj + 256, 320, 64, 256, nullptr, tmpf, 256,
       xbuf, nullptr, 256, 0, N_NODES);

  // attention layers: fused QKV projection, gather-after-project attention, out-proj+ReLU
  for (int l = 0; l < NLAYERS; ++l) {
    gemm(xbuf, 256, nullptr, wbf_inp + (size_t)l * 768 * 256, 256, 256, 768,
         in_proj_b + l * 768, nullptr, 0, qkv, nullptr, 768, 0, N_NODES);
    attn_ctx<<<N_NODES / 8, 256, 0, stream>>>(qkv, ctx_idx, kpm, obuf);
    gemm(obuf, 256, nullptr, wbf_out + (size_t)l * 256 * 256, 256, 256, 256,
         out_b + l * 256, nullptr, 0, xbuf, nullptr, 256, 1, N_NODES);
  }

  // edge MLP: h1 = relu(x[u]@W1a^T + x[v]@W1b^T + ef@W1c^T + b1)
  edge_c0<<<(NEDGES * 256 + 255) / 256, 256, 0, stream>>>(edge_feats, mlp_w1, mlp_b1, tmpf);
  gemm(xbuf, 256, u_idx, wbf_m1,       514, 256, 256, nullptr, tmpf, 256,
       nullptr, tmpf, 256, 0, NEDGES);
  gemm(xbuf, 256, v_idx, wbf_m1 + 256, 514, 256, 256, nullptr, tmpf, 256,
       h1, nullptr, 256, 1, NEDGES);
  // h2 = relu(h1 @ W2^T + b2)
  gemm(h1, 256, nullptr, wbf_m2, 256, 256, 128, mlp_b2, nullptr, 0,
       h2, nullptr, 128, 1, NEDGES);
  // logits
  edge_logits<<<NEDGES / 256, 256, 0, stream>>>(h2, mlp_w3, mlp_b3, logits);
}
